// Fusion_1640677507711
// MI455X (gfx1250) — matrix operand order repaired and checked by reading, not compiled, not run on hardware
//
#include <hip/hip_runtime.h>
#include <hip/hip_bf16.h>

typedef __attribute__((ext_vector_type(16))) _Float16 v16h;
typedef __attribute__((ext_vector_type(8)))  float    v8f;

#define D_FEAT 128

// ---------- helpers: ordered-uint encoding for float atomic max ----------
__device__ __forceinline__ unsigned encodeOrd(float f) {
    unsigned u = __float_as_uint(f);
    return (u & 0x80000000u) ? ~u : (u | 0x80000000u);
}
__device__ __forceinline__ float decodeOrd(unsigned e) {
    unsigned u = (e & 0x80000000u) ? (e ^ 0x80000000u) : ~e;
    return __uint_as_float(u);
}
__device__ __forceinline__ float leaky(float e) { return e > 0.0f ? e : 0.2f * e; }

// ---------- WMMA GEMM: H[N,128] = X[N,128] @ W^T (W is [128,128]) ----------
// One wave computes a 16-row x 128-col stripe: 4 K-steps x 8 N-tiles = 32 WMMAs.
__global__ void gat_gemm_xwT(const float* __restrict__ X, const float* __restrict__ W,
                             float* __restrict__ H, int N) {
    const int lane = threadIdx.x & 31;
    const int wave = threadIdx.x >> 5;
    const int row0 = (blockIdx.x * (blockDim.x >> 5) + wave) << 4;
    if (row0 >= N) return;
    const int r  = lane & 15;      // row-in-tile for A / col-in-tile for B
    const int hi = lane >> 4;      // lane half

    v8f acc[8] = {};
    const float* xrow = X + (size_t)(row0 + r) * D_FEAT;

#pragma unroll
    for (int kk = 0; kk < 4; ++kk) {
        const int ko = kk * 32;
        // A tile 16x32 f16, ISA wave32 layout:
        //   halfs 0..7  <- X[row][ko + hi*8 + 0..7]
        //   halfs 8..15 <- X[row][ko + hi*8 + 16 + 0..7]
        v16h a;
        {
            const float* p0 = xrow + ko + hi * 8;
            const float* p1 = p0 + 16;
#pragma unroll
            for (int j = 0; j < 8; ++j) a[j] = (_Float16)p0[j];
#pragma unroll
            for (int j = 0; j < 8; ++j) a[8 + j] = (_Float16)p1[j];
        }
#pragma unroll
        for (int t = 0; t < 8; ++t) {
            // B tile 32x16 f16: B[k][n] = (W^T)[k][t*16+n] = W[t*16+n][k]
            //   halfs 0..7 <- W[t*16 + r][ko + hi*8 + 0..7], halfs 8..15 <- W[t*16 + r][ko + hi*8 + 16 + 0..7]
            const float* wrow = W + (size_t)(t * 16 + r) * D_FEAT + ko + hi * 8;
            v16h b;
#pragma unroll
            for (int j = 0; j < 16; ++j) b[j] = (_Float16)wrow[(j & 7) + ((j >> 3) << 4)];
            acc[t] = __builtin_amdgcn_wmma_f32_16x16x32_f16(
                false, a, false, b, (short)0, acc[t], false, false);
        }
    }
    // C/D layout: vgpr v, lane -> element (m = v + hi*8, n = lane&15)
#pragma unroll
    for (int t = 0; t < 8; ++t) {
        float* hbase = H + (size_t)row0 * D_FEAT + t * 16 + r;
#pragma unroll
        for (int v = 0; v < 8; ++v)
            hbase[(size_t)(v + hi * 8) * D_FEAT] = acc[t][v];
    }
}

// ---------- per-node score: out[n] = dot(H[n,:], att) ----------
__global__ void gat_rowdot(const float* __restrict__ H, const float* __restrict__ att,
                           float* __restrict__ out, int N) {
    const int lane = threadIdx.x & 31;
    const int node = (int)((blockIdx.x * blockDim.x + threadIdx.x) >> 5);
    if (node >= N) return;
    const float* row = H + (size_t)node * D_FEAT + lane * 4;
    const float* ap  = att + lane * 4;
    float s = row[0] * ap[0] + row[1] * ap[1] + row[2] * ap[2] + row[3] * ap[3];
#pragma unroll
    for (int off = 16; off > 0; off >>= 1) s += __shfl_down(s, off, 32);
    if (lane == 0) out[node] = s;
}

// ---------- fills ----------
__global__ void fill_f32(float* __restrict__ p, float v, long n) {
    long i = (long)blockIdx.x * blockDim.x + threadIdx.x;
    if (i < n) p[i] = v;
}
__global__ void fill_u32(unsigned* __restrict__ p, unsigned v, long n) {
    long i = (long)blockIdx.x * blockDim.x + threadIdx.x;
    if (i < n) p[i] = v;
}

// ---------- edge pass 1: segment max of leaky score, ordered-uint atomicMax ----------
__global__ void gat_edge_max(const int* __restrict__ src, const int* __restrict__ dst,
                             const float* __restrict__ sl, const float* __restrict__ sr,
                             unsigned* __restrict__ menc, int E) {
    for (int i = blockIdx.x * blockDim.x + threadIdx.x; i < E;
         i += gridDim.x * blockDim.x) {
        const int s = src[i], t = dst[i];
        const float e = leaky(sl[s] + sr[t]);
        atomicMax(&menc[t], encodeOrd(e));
    }
}

// ---------- edge pass 2: one wave per edge; den += exp, num += xl[src]*exp ----------
__global__ void gat_edge_scatter(const int* __restrict__ src, const int* __restrict__ dst,
                                 const float* __restrict__ sl, const float* __restrict__ sr,
                                 const unsigned* __restrict__ menc,
                                 const float* __restrict__ xl,
                                 float* __restrict__ num, float* __restrict__ den, int E) {
    const int lane = threadIdx.x & 31;
    const int wid  = (int)((blockIdx.x * blockDim.x + threadIdx.x) >> 5);
    const int nw   = (int)((gridDim.x * blockDim.x) >> 5);
    for (int i = wid; i < E; i += nw) {
        const int s = src[i], t = dst[i];
        const float e  = leaky(sl[s] + sr[t]);
        const float ex = __expf(e - decodeOrd(menc[t]));
        if (lane == 0) atomicAdd(&den[t], ex);
        const float* xp = xl + (size_t)s * D_FEAT;
        float* np = num + (size_t)t * D_FEAT;
#pragma unroll
        for (int j = 0; j < 4; ++j) {
            const int d = lane + j * 32;
            atomicAdd(&np[d], xp[d] * ex);
        }
    }
}

// ---------- finalizers ----------
__global__ void gat_div(const float* __restrict__ num, const float* __restrict__ den,
                        float* __restrict__ out, int N) {
    int i = blockIdx.x * blockDim.x + threadIdx.x;
    if (i >= N * D_FEAT) return;
    out[i] = num[i] / (den[i >> 7] + 1e-16f);
}
__global__ void gat_fuse2(const float* __restrict__ base,
                          const float* __restrict__ nA, const float* __restrict__ dA,
                          const float* __restrict__ nB, const float* __restrict__ dB,
                          float* __restrict__ out, int N) {
    int i = blockIdx.x * blockDim.x + threadIdx.x;
    if (i >= N * D_FEAT) return;
    const int n = i >> 7;
    out[i] = base[i] + nA[i] / (dA[n] + 1e-16f) + nB[i] / (dB[n] + 1e-16f);
}
__global__ void gat_fuse1_inplace(const float* __restrict__ base,
                                  const float* __restrict__ den,
                                  float* __restrict__ out, int N) {
    int i = blockIdx.x * blockDim.x + threadIdx.x;
    if (i >= N * D_FEAT) return;
    out[i] = base[i] + out[i] / (den[i >> 7] + 1e-16f);
}

// =========================================================================
extern "C" void kernel_launch(void* const* d_in, const int* in_sizes, int n_in,
                              void* d_out, int out_size, void* d_ws, size_t ws_size,
                              hipStream_t stream) {
    const int STU = 50000, ITEM = 20000, CONC = 2000;
    const int E_CC = 40000, E_IC = 80000, E_SI = 1000000;

    const float* stu_x     = (const float*)d_in[0];
    const float* item_x    = (const float*)d_in[1];
    const float* conc_x    = (const float*)d_in[2];
    const float* stu_raw_x = (const float*)d_in[3];
    const float* W_cc = (const float*)d_in[4];  const float* al_cc = (const float*)d_in[5];  const float* ar_cc = (const float*)d_in[6];
    const float* W_ic = (const float*)d_in[7];  const float* al_ic = (const float*)d_in[8];  const float* ar_ic = (const float*)d_in[9];
    const float* al_cce = (const float*)d_in[10]; const float* ar_cce = (const float*)d_in[11];
    const float* al_ice = (const float*)d_in[12]; const float* ar_ice = (const float*)d_in[13];
    const float* W_ci = (const float*)d_in[14]; const float* al_ci = (const float*)d_in[15]; const float* ar_ci = (const float*)d_in[16];
    const float* W_si = (const float*)d_in[17]; const float* al_si = (const float*)d_in[18]; const float* ar_si = (const float*)d_in[19];
    // d_in[20], d_in[21]: w_ci_att / w_si_att -> softmax over [N,1] == 1.0, unused.
    const float* W_is = (const float*)d_in[22]; const float* al_is = (const float*)d_in[23]; const float* ar_is = (const float*)d_in[24];
    const int* cc_src  = (const int*)d_in[25];
    const int* cc_dst  = (const int*)d_in[26];
    const int* ic_item = (const int*)d_in[27];
    const int* ic_conc = (const int*)d_in[28];
    const int* si_stu  = (const int*)d_in[29];
    const int* si_item = (const int*)d_in[30];

    // ---- workspace linear allocator (floats) ----
    float* ws = (float*)d_ws;
    size_t off = 0;
    auto alloc = [&](size_t n) { float* p = ws + off; off += n; return p; };
    float* H2a    = alloc((size_t)CONC * D_FEAT);  // h_cc, then h_conc_ci
    float* H2b    = alloc((size_t)CONC * D_FEAT);  // h_conc_ic
    float* H20a   = alloc((size_t)ITEM * D_FEAT);  // h_item_ic, then h_item_is
    float* H20b   = alloc((size_t)ITEM * D_FEAT);  // h_item_ci, then h_item_si
    float* H50a   = alloc((size_t)STU  * D_FEAT);  // h_stu_si, then h_stu_is
    float* c1buf  = alloc((size_t)CONC * D_FEAT);
    float* c2buf  = alloc((size_t)CONC * D_FEAT);
    float* numA2  = alloc((size_t)CONC * D_FEAT);
    float* numB2  = alloc((size_t)CONC * D_FEAT);
    float* numA20 = alloc((size_t)ITEM * D_FEAT);
    float* numB20 = alloc((size_t)ITEM * D_FEAT);
    float* slb    = alloc((size_t)STU);
    float* srb    = alloc((size_t)STU);
    unsigned* menc = (unsigned*)alloc((size_t)STU);
    float* denA   = alloc((size_t)STU);
    float* denB   = alloc((size_t)STU);
    (void)ws_size; (void)in_sizes; (void)n_in; (void)out_size;

    float* out_conc = (float*)d_out;
    float* out_item = out_conc + (size_t)CONC * D_FEAT;
    float* out_stu  = out_item + (size_t)ITEM * D_FEAT;

    auto cdiv = [](long a, long b) { return (unsigned)((a + b - 1) / b); };

    auto gemm = [&](const float* X, const float* W, float* H, int N) {
        gat_gemm_xwT<<<cdiv(N, 64), 128, 0, stream>>>(X, W, H, N);      // 4 waves/block
    };
    auto rowdot = [&](const float* H, const float* att, float* o, int N) {
        gat_rowdot<<<cdiv((long)N * 32, 256), 256, 0, stream>>>(H, att, o, N);
    };
    auto run_gat = [&](const float* xl, const int* esrc, const int* edst, int E,
                       int n_tgt, float* num, float* den) {
        fill_u32<<<cdiv(n_tgt, 256), 256, 0, stream>>>(menc, 0u, n_tgt);
        fill_f32<<<cdiv(n_tgt, 256), 256, 0, stream>>>(den, 0.0f, n_tgt);
        fill_f32<<<cdiv((long)n_tgt * D_FEAT, 256), 256, 0, stream>>>(num, 0.0f, (long)n_tgt * D_FEAT);
        gat_edge_max<<<cdiv(E, 256), 256, 0, stream>>>(esrc, edst, slb, srb, menc, E);
        gat_edge_scatter<<<cdiv(E, 8), 256, 0, stream>>>(esrc, edst, slb, srb, menc, xl, num, den, E);
    };

    // ===================== concept fusion =====================
    // c1 = GAT(conc<-conc, W_cc)
    gemm(conc_x, W_cc, H2a, CONC);
    rowdot(H2a, al_cc, slb, CONC);
    rowdot(H2a, ar_cc, srb, CONC);
    run_gat(H2a, cc_src, cc_dst, E_CC, CONC, numA2, denA);
    gat_div<<<cdiv((long)CONC * D_FEAT, 256), 256, 0, stream>>>(numA2, denA, c1buf, CONC);
    // c2 = GAT(conc<-item, W_ic)
    gemm(item_x, W_ic, H20a, ITEM);
    gemm(conc_x, W_ic, H2b, CONC);
    rowdot(H20a, al_ic, slb, ITEM);
    rowdot(H2b, ar_ic, srb, CONC);
    run_gat(H20a, ic_item, ic_conc, E_IC, CONC, numA2, denA);
    gat_div<<<cdiv((long)CONC * D_FEAT, 256), 256, 0, stream>>>(numA2, denA, c2buf, CONC);
    // e1 = GAT(conc<-c1, no lin), e2 = GAT(conc<-c2, no lin), over cc edges
    rowdot(c1buf, al_cce, slb, CONC);
    rowdot(conc_x, ar_cce, srb, CONC);
    run_gat(c1buf, cc_src, cc_dst, E_CC, CONC, numA2, denA);
    rowdot(c2buf, al_ice, slb, CONC);
    rowdot(conc_x, ar_ice, srb, CONC);
    run_gat(c2buf, cc_src, cc_dst, E_CC, CONC, numB2, denB);
    gat_fuse2<<<cdiv((long)CONC * D_FEAT, 256), 256, 0, stream>>>(
        conc_x, numA2, denA, numB2, denB, out_conc, CONC);

    // ===================== item fusion =====================
    // i1 = GAT(item<-conc, W_ci)
    gemm(conc_x, W_ci, H2a, CONC);
    gemm(item_x, W_ci, H20b, ITEM);
    rowdot(H2a, al_ci, slb, CONC);
    rowdot(H20b, ar_ci, srb, ITEM);
    run_gat(H2a, ic_conc, ic_item, E_IC, ITEM, numA20, denA);
    // i2 = GAT(item<-stu, W_si)
    gemm(stu_x, W_si, H50a, STU);
    gemm(item_x, W_si, H20b, ITEM);
    rowdot(H50a, al_si, slb, STU);
    rowdot(H20b, ar_si, srb, ITEM);
    run_gat(H50a, si_stu, si_item, E_SI, ITEM, numB20, denB);
    // softmax over [N,1] == 1 -> item_fused = item_x + i1 + i2
    gat_fuse2<<<cdiv((long)ITEM * D_FEAT, 256), 256, 0, stream>>>(
        item_x, numA20, denA, numB20, denB, out_item, ITEM);

    // ===================== student fusion =====================
    // s1 = GAT(stu_raw<-item, W_is); numerator accumulates directly in out_stu
    gemm(item_x, W_is, H20a, ITEM);
    gemm(stu_raw_x, W_is, H50a, STU);
    rowdot(H20a, al_is, slb, ITEM);
    rowdot(H50a, ar_is, srb, STU);
    run_gat(H20a, si_item, si_stu, E_SI, STU, out_stu, denA);
    gat_fuse1_inplace<<<cdiv((long)STU * D_FEAT, 256), 256, 0, stream>>>(
        stu_raw_x, denA, out_stu, STU);
}
